// GraphMRConv_11347303596504
// MI455X (gfx1250) — matrix operand order repaired and checked by reading, not compiled, not run on hardware
//
#include <hip/hip_runtime.h>
#include <math.h>

#define BATCH 8
#define NNODE 16384
#define CH    128
#define OUTC  256
#define NEDGE 262144

typedef __attribute__((ext_vector_type(16))) _Float16 v16h;
typedef __attribute__((ext_vector_type(8)))  float    v8f;

// Order-preserving float -> uint encoding so integer atomicMax == float max.
__device__ __forceinline__ unsigned enc_f32(float f) {
    unsigned u = __float_as_uint(f);
    return (u & 0x80000000u) ? ~u : (u | 0x80000000u);
}
__device__ __forceinline__ float dec_key(unsigned k) {
    // k==0 means "no incoming edge" -> reference maps non-finite to 0.0
    if (k == 0u) return 0.0f;
    unsigned u = (k & 0x80000000u) ? (k & 0x7FFFFFFFu) : ~k;
    return __uint_as_float(u);
}

// ---------------------------------------------------------------------------
// Kernel 1: per-edge scatter max.  One wave (32 lanes) handles one (batch,
// edge): lanes cover 128 channels as float4 -> 512B coalesced gather from
// x[col] and 512B of u32 atomicMax into keys[row].  All traffic is L2-resident
// (8MB x-slab, 8MB key-slab per batch).
// ---------------------------------------------------------------------------
__global__ __launch_bounds__(256)
void scatter_max_kernel(const float* __restrict__ x,
                        const long long* __restrict__ ei,
                        unsigned* __restrict__ keys) {
    __shared__ int srow[8];
    __shared__ int scol[8];

    const unsigned blk = blockIdx.x;          // 262144 blocks total
    const int bb     = blk >> 15;             // 32768 blocks per batch
    const int e_base = (blk & 32767) * 8;     // 8 edges per block

    if (threadIdx.x < 8) {
        const int e = e_base + threadIdx.x;
        srow[threadIdx.x] = (int)ei[e];          // row = edge_index[0]
        scol[threadIdx.x] = (int)ei[NEDGE + e];  // col = edge_index[1]
    }
    __syncthreads();

    const int el   = threadIdx.x >> 5;        // wave id = local edge
    const int lane = threadIdx.x & 31;
    const int row  = srow[el];
    const int col  = scol[el];

    const size_t src = ((size_t)bb * NNODE + col) * CH + lane * 4;
    const size_t dst = ((size_t)bb * NNODE + row) * CH + lane * 4;

    const float4 xv = *(const float4*)(x + src);
    unsigned* d = keys + dst;
    atomicMax(d + 0, enc_f32(xv.x));
    atomicMax(d + 1, enc_f32(xv.y));
    atomicMax(d + 2, enc_f32(xv.z));
    atomicMax(d + 3, enc_f32(xv.w));
}

// ---------------------------------------------------------------------------
// Kernel 2: fused (dec(keys)-x) @ W + b -> exact GELU, via WMMA f16->f32.
// Block = 256 threads (8 waves), covers 64 rows x 128 out-cols.
//   wave w: row-tile rt = w>>1 (4 tiles of 16), col-group cg = w&1 (64 cols)
//   per wave: 4 K-steps x 4 col-tiles = 16 x v_wmma_f32_16x16x32_f16
// LDS: A-tile 64x128 f16 (stride 136) + W-tile 128x128 f16 transposed
// (stride 136) = 52KB.  Stride 136 halves = 272B = 68 dwords -> b128 fragment
// loads from 16 lanes hit 16 disjoint bank groups (no LDS conflicts).
// ---------------------------------------------------------------------------
__global__ __launch_bounds__(256)
void gemm_gelu_kernel(const unsigned* __restrict__ keys,
                      const float* __restrict__ x,
                      const float* __restrict__ W,
                      const float* __restrict__ bias,
                      float* __restrict__ out) {
    __shared__ __align__(16) _Float16 lds_a[64][136];   // [row][k]
    __shared__ __align__(16) _Float16 lds_w[128][136];  // [o_local][k] (W^T)

    const int tid      = threadIdx.x;
    const int row_base = blockIdx.x * 64;       // 2048 blocks in x
    const int col_base = blockIdx.y * 128;      // 2 blocks in y

    // ---- Stage A = dec(keys) - x as f16 (64 rows x 128 ch) ----
    #pragma unroll
    for (int i = tid; i < 64 * CH / 4; i += 256) {
        const int r  = i >> 5;          // 32 float4 per 128-ch row
        const int c4 = i & 31;
        const size_t g = (size_t)(row_base + r);
        const uint4  kv = ((const uint4*)(keys + g * CH))[c4];
        const float4 xv = ((const float4*)(x    + g * CH))[c4];
        const int c = c4 * 4;
        lds_a[r][c + 0] = (_Float16)(dec_key(kv.x) - xv.x);
        lds_a[r][c + 1] = (_Float16)(dec_key(kv.y) - xv.y);
        lds_a[r][c + 2] = (_Float16)(dec_key(kv.z) - xv.z);
        lds_a[r][c + 3] = (_Float16)(dec_key(kv.w) - xv.w);
    }

    // ---- Stage W^T as f16: lds_w[o_local][k], o = col_base + o_local ----
    #pragma unroll
    for (int i = tid; i < CH * 128 / 4; i += 256) {
        const int k  = i >> 5;          // 32 float4 per 128-col slice
        const int ol = (i & 31) * 4;
        const float4 wv = *(const float4*)(W + (size_t)k * OUTC + col_base + ol);
        lds_w[ol + 0][k] = (_Float16)wv.x;
        lds_w[ol + 1][k] = (_Float16)wv.y;
        lds_w[ol + 2][k] = (_Float16)wv.z;
        lds_w[ol + 3][k] = (_Float16)wv.w;
    }

    __syncthreads();

    const int w    = tid >> 5;
    const int lane = tid & 31;
    const int rt   = w >> 1;            // 0..3 -> rows rt*16..rt*16+15
    const int cg   = w & 1;             // 0..1 -> cols cg*64..cg*64+63
    const int hi   = lane >> 4;
    const int ln   = lane & 15;
    const int arow = rt * 16 + ln;

    union Frag { uint4 q[2]; v16h h; };

    v8f acc[4] = {};

    #pragma unroll
    for (int kb = 0; kb < 4; ++kb) {
        const int kbase = kb * 32;
        // A 16x32 f16 layout: lanes 0-15 -> M=ln; VGPR0-3 K=kbase+hi*8+{0..7},
        // VGPR4-7 K=kbase+hi*8+16+{0..7}  -> two b128 LDS loads.
        Frag a;
        a.q[0] = *(const uint4*)&lds_a[arow][kbase + hi * 8];
        a.q[1] = *(const uint4*)&lds_a[arow][kbase + hi * 8 + 16];

        #pragma unroll
        for (int ct = 0; ct < 4; ++ct) {
            const int n = cg * 64 + ct * 16 + ln;
            // B 32x16 f16 layout: lane column = ln; VGPR0-3 K=kbase+hi*8+{0..7},
            // VGPR4-7 K=kbase+hi*8+16+{0..7} -> two b128 LDS loads.
            Frag b;
            b.q[0] = *(const uint4*)&lds_w[n][kbase + hi * 8];
            b.q[1] = *(const uint4*)&lds_w[n][kbase + hi * 8 + 16];

            acc[ct] = __builtin_amdgcn_wmma_f32_16x16x32_f16(
                false, a.h, false, b.h, (short)0, acc[ct], false, false);
        }
    }

    // ---- Epilogue: bias + exact (erf) GELU, store f32 C/D layout ----
    #pragma unroll
    for (int ct = 0; ct < 4; ++ct) {
        const int o  = col_base + cg * 64 + ct * 16 + ln;
        const float bv = bias[o];
        #pragma unroll
        for (int r = 0; r < 8; ++r) {
            const size_t g = (size_t)(row_base + rt * 16 + hi * 8 + r);
            const float h = acc[ct][r] + bv;
            out[g * OUTC + o] = 0.5f * h * (1.0f + erff(h * 0.70710678118654752f));
        }
    }
}

// ---------------------------------------------------------------------------
extern "C" void kernel_launch(void* const* d_in, const int* in_sizes, int n_in,
                              void* d_out, int out_size, void* d_ws, size_t ws_size,
                              hipStream_t stream) {
    (void)in_sizes; (void)n_in; (void)out_size; (void)ws_size;

    const float*     x    = (const float*)d_in[0];
    const long long* ei   = (const long long*)d_in[1];   // int64 (2, E)
    const float*     W    = (const float*)d_in[2];
    const float*     bias = (const float*)d_in[3];
    float*           out  = (float*)d_out;
    unsigned*        keys = (unsigned*)d_ws;             // 64 MB of scratch

    const size_t key_bytes = (size_t)BATCH * NNODE * CH * sizeof(unsigned);
    hipMemsetAsync(keys, 0, key_bytes, stream);          // key 0 == -inf / empty

    // 8 batches * 262144 edges, one wave per edge -> 262144 blocks of 256.
    scatter_max_kernel<<<dim3(262144), dim3(256), 0, stream>>>(x, ei, keys);

    // 131072 rows / 64 per block; 256 cols / 128 per block.
    gemm_gelu_kernel<<<dim3(2048, 2), dim3(256), 0, stream>>>(keys, x, W, bias, out);
}
